// SingleStreamBlock_26156350833173
// MI455X (gfx1250) — matrix operand order repaired and checked by reading, not compiled, not run on hardware
//
#include <hip/hip_runtime.h>
#include <hip/hip_bf16.h>
#include <math.h>

// ---------------- problem constants ----------------
#define Lq 2048
#define Dm 3072
#define Hh 24
#define DH 128
#define MLPD 12288
#define N1 21504            // 3*D + MLP
#define K2 15360            // D + MLP
#define MODN 9216           // 3*D

typedef __attribute__((ext_vector_type(16))) __bf16 v16bf;
typedef __attribute__((ext_vector_type(8)))  float  v8f;
typedef __attribute__((ext_vector_type(4)))  unsigned int u32x4;
typedef __attribute__((ext_vector_type(8)))  int          i32x8;
typedef __attribute__((ext_vector_type(4)))  int          i32x4;

union Frag16 { v16bf v; uint4 u[2]; unsigned short s[16]; };

static __device__ __forceinline__ v8f vzero() {
  v8f z;
#pragma unroll
  for (int i = 0; i < 8; ++i) z[i] = 0.f;
  return z;
}

static __device__ __forceinline__ v8f wmma_bf16(v16bf a, v16bf b, v8f c) {
  // D = A*B + C, 16x16x32 bf16 -> f32
  return __builtin_amdgcn_wmma_f32_16x16x32_bf16(false, a, false, b, (short)0, c, false, false);
}

static __device__ __forceinline__ unsigned short f2bf(float f) {
  unsigned int u = __float_as_uint(f);
  unsigned int r = u + 0x7FFFu + ((u >> 16) & 1u);   // RNE
  return (unsigned short)(r >> 16);
}
static __device__ __forceinline__ float bf2f(unsigned short s) {
  return __uint_as_float(((unsigned int)s) << 16);
}

// ---- TDM: async 2D tile load (global bf16 -> LDS), rows x cols, row stride in elements ----
static __device__ __forceinline__ void tdm_load_tile_bf16(
    const unsigned short* gsrc, unsigned int lds_byte_addr,
    unsigned int tile_cols, unsigned int tile_rows,
    unsigned int row_stride_elems) {
  unsigned long long ga = (unsigned long long)(uintptr_t)gsrc;

  u32x4 g0;
  g0[0] = 1u;                                   // count=1, user descriptor
  g0[1] = lds_byte_addr;                        // [63:32] lds_addr
  g0[2] = (unsigned int)ga;                     // [95:64] global_addr lo
  g0[3] = ((unsigned int)(ga >> 32) & 0x01FFFFFFu) | 0x80000000u; // addr hi + type=2

  unsigned int td0 = row_stride_elems;          // tensor_dim0 (elements per full row)
  unsigned int td1 = tile_rows;                 // tensor_dim1 (rows valid from tile start)
  unsigned long long st0 = (unsigned long long)row_stride_elems; // tensor_dim0_stride

  i32x8 g1;
  g1[0] = (int)(1u << 16);                      // workgroup_mask=0, data_size=1 (2 bytes)
  g1[1] = (int)((td0 & 0xFFFFu) << 16);         // [47:32]=atomic addr(0), [63:48]=td0 lo
  g1[2] = (int)(((td0 >> 16) & 0xFFFFu) | ((td1 & 0xFFFFu) << 16));
  g1[3] = (int)(((td1 >> 16) & 0xFFFFu) | (tile_cols << 16));     // [127:112] tile_dim0
  g1[4] = (int)(tile_rows & 0xFFFFu);           // [143:128] tile_dim1, tile_dim2=0
  g1[5] = (int)(unsigned int)(st0 & 0xFFFFFFFFu);
  g1[6] = (int)(unsigned int)((st0 >> 32) & 0xFFFFu);             // stride0 hi, stride1 lo=0
  g1[7] = 0;

  i32x4 z4; z4[0] = 0; z4[1] = 0; z4[2] = 0; z4[3] = 0;
#if __clang_major__ >= 23
  i32x8 z8;
#pragma unroll
  for (int i = 0; i < 8; ++i) z8[i] = 0;
  __builtin_amdgcn_tensor_load_to_lds(g0, g1, z4, z4, z8, 0);
#else
  __builtin_amdgcn_tensor_load_to_lds(g0, g1, z4, z4, 0);
#endif
}

// ---------------- K0: fp32 weights -> bf16 (once per call; bf16 copy is L2-resident) ----
__global__ void k_cvt_w(const float* __restrict__ w, unsigned short* __restrict__ o) {
  size_t i = ((size_t)blockIdx.x * 256 + threadIdx.x) * 4;
  float4 v = *(const float4*)&w[i];
  unsigned short r[4] = { f2bf(v.x), f2bf(v.y), f2bf(v.z), f2bf(v.w) };
  *(uint2*)&o[i] = *(const uint2*)r;
}

// ---------------- K1: mod = silu(vec) @ mod_w + mod_b ----------------
__global__ void k_mod_gemv(const float* __restrict__ vec, const float* __restrict__ mod_w,
                           const float* __restrict__ mod_b, float* __restrict__ mod) {
  __shared__ float sv[Dm];
  for (int i = threadIdx.x; i < Dm; i += 256) {
    float v = vec[i];
    sv[i] = v / (1.f + __expf(-v));
  }
  __syncthreads();
  int col = blockIdx.x * 256 + threadIdx.x;   // 36 blocks
  float acc = 0.f;
  for (int i = 0; i < Dm; ++i) acc = fmaf(sv[i], mod_w[(size_t)i * MODN + col], acc);
  mod[col] = acc + mod_b[col];
}

// ---------------- K2: layernorm + modulate -> bf16 ----------------
__global__ void k_ln_mod(const float* __restrict__ x, const float* __restrict__ mod,
                         unsigned short* __restrict__ xmod) {
  int row = blockIdx.x;
  const float* xr = x + (size_t)row * Dm;
  __shared__ float rs[256], rs2[256];
  float s = 0.f, s2 = 0.f;
  for (int i = threadIdx.x; i < Dm; i += 256) {
    float v = xr[i];
    s += v; s2 += v * v;
  }
  rs[threadIdx.x] = s; rs2[threadIdx.x] = s2;
  __syncthreads();
  for (int t = 128; t > 0; t >>= 1) {
    if (threadIdx.x < t) { rs[threadIdx.x] += rs[threadIdx.x + t]; rs2[threadIdx.x] += rs2[threadIdx.x + t]; }
    __syncthreads();
  }
  float mu = rs[0] * (1.f / Dm);
  float var = rs2[0] * (1.f / Dm) - mu * mu;
  float rinv = rsqrtf(var + 1e-6f);
  for (int i = threadIdx.x; i < Dm; i += 256) {
    float xn = (xr[i] - mu) * rinv;
    float y  = (1.f + mod[Dm + i]) * xn + mod[i];
    xmod[(size_t)row * Dm + i] = f2bf(y);
  }
}

// ---------------- GEMM: C = A(bf16 MxK) * B(bf16 KxN) ----------------
// BM=BN=128, BK=32, 256 threads = 8 waves; wave w owns a 16x128 strip.
// A tiles staged by the Tensor Data Mover (double-buffered, wave 0 issues,
// s_wait_tensorcnt(1) keeps the k+1 prefetch in flight while k is consumed).
// mode 0: outbf = bf16(acc + bias)          (h = x_mod @ lin1_w + b)
// mode 1: outf  = (acc + bias)*gate + resid (final output)
__global__ void k_gemm(const unsigned short* __restrict__ A, const unsigned short* __restrict__ B,
                       const float* __restrict__ bias, int N, int K, int mode,
                       unsigned short* __restrict__ outbf, float* __restrict__ outf,
                       const float* __restrict__ gate, const float* __restrict__ resid) {
  __shared__ __align__(16) unsigned short As[2][128 * 32];    // [row][k]
  __shared__ __align__(16) unsigned short BsT[2][128 * 32];   // [n][k] (transposed)

  const int tid  = threadIdx.x;
  const int w    = tid >> 5;
  const int lane = tid & 31;
  const int m    = lane & 15;
  const int hi   = (lane & 16) ? 1 : 0;
  const int kbA  = hi ? 8 : 0;     // A-frag K base (interleaved layout)
  const int kbB  = hi ? 8 : 0;     // B-frag K base (interleaved layout)
  const int rowBase = blockIdx.y * 128;
  const int colBase = blockIdx.x * 128;

  v8f acc[8];
#pragma unroll
  for (int i = 0; i < 8; ++i) acc[i] = vzero();

  const int nk = K / 32;

  auto issueA = [&](int kt, int buf) {
    tdm_load_tile_bf16(A + (size_t)rowBase * K + (size_t)kt * 32,
                       (unsigned int)(uintptr_t)(void*)&As[buf][0],
                       32u, 128u, (unsigned int)K);
  };
  auto stageB = [&](int kt, int buf) {
    for (int c = tid; c < 512; c += 256) {          // 32k x 128n = 512 chunks of 8 bf16
      int k = c >> 4, n = (c & 15) * 8;
      uint4 d = *(const uint4*)&B[(size_t)(kt * 32 + k) * N + colBase + n];
      const unsigned short* sp = (const unsigned short*)&d;
#pragma unroll
      for (int j = 0; j < 8; ++j) BsT[buf][(n + j) * 32 + k] = sp[j];
    }
  };

  // prologue: tile 0 in flight
  if (w == 0) issueA(0, 0);
  stageB(0, 0);

  for (int kt = 0; kt < nk; ++kt) {
    const int cur = kt & 1, nxt = cur ^ 1;
    const bool more = (kt + 1 < nk);
    if (more) {
      if (w == 0) issueA(kt + 1, nxt);
      stageB(kt + 1, nxt);
      __builtin_prefetch(&B[(size_t)((kt + 1) * 32) * N + colBase], 0, 1);
    }
    if (w == 0) {
      if (more) __builtin_amdgcn_s_wait_tensorcnt(1);   // tile kt resident, kt+1 in flight
      else      __builtin_amdgcn_s_wait_tensorcnt(0);
    }
    __syncthreads();

    Frag16 fa;
    fa.u[0] = *(const uint4*)&As[cur][(w * 16 + m) * 32 + kbA];
    fa.u[1] = *(const uint4*)&As[cur][(w * 16 + m) * 32 + kbA + 16];
    Frag16 fb[8];
#pragma unroll
    for (int n8 = 0; n8 < 8; ++n8) {
      fb[n8].u[0] = *(const uint4*)&BsT[cur][(n8 * 16 + m) * 32 + kbB];
      fb[n8].u[1] = *(const uint4*)&BsT[cur][(n8 * 16 + m) * 32 + kbB + 16];
    }
#pragma unroll
    for (int n8 = 0; n8 < 8; ++n8)
      acc[n8] = wmma_bf16(fa.v, fb[n8].v, acc[n8]);
    __syncthreads();
  }

  // epilogue: C layout -> row = base + w*16 + v + 8*hi, col = base + n8*16 + m
#pragma unroll
  for (int n8 = 0; n8 < 8; ++n8) {
    int col = colBase + n8 * 16 + m;
    float b = bias[col];
#pragma unroll
    for (int v = 0; v < 8; ++v) {
      int row = rowBase + w * 16 + v + 8 * hi;
      float val = acc[n8][v] + b;
      if (mode == 0) {
        outbf[(size_t)row * N + col] = f2bf(val);
      } else {
        outf[(size_t)row * Dm + col] = val * gate[col] + resid[(size_t)row * Dm + col];
      }
    }
  }
}

// ---------------- K4: q/k RMSNorm + RoPE, v repack; head-major bf16 ----------------
__global__ void k_qkv_post(const unsigned short* __restrict__ h, const float* __restrict__ pe,
                           const float* __restrict__ qsc, const float* __restrict__ ksc,
                           unsigned short* __restrict__ qr, unsigned short* __restrict__ kr,
                           unsigned short* __restrict__ vh) {
  int l = blockIdx.x, hd = blockIdx.y, d = threadIdx.x;  // 128 threads
  const unsigned short* hp = h + (size_t)l * N1;
  float q = bf2f(hp[hd * DH + d]);
  float k = bf2f(hp[Dm + hd * DH + d]);
  float v = bf2f(hp[2 * Dm + hd * DH + d]);

  __shared__ float sq[128], sk[128], sp[128];
  sq[d] = q * q; sk[d] = k * k;
  __syncthreads();
  for (int t = 64; t > 0; t >>= 1) {
    if (d < t) { sq[d] += sq[d + t]; sk[d] += sk[d + t]; }
    __syncthreads();
  }
  float qn = q * rsqrtf(sq[0] * (1.f / DH) + 1e-6f) * qsc[d];
  float kn = k * rsqrtf(sk[0] * (1.f / DH) + 1e-6f) * ksc[d];
  __syncthreads();

  // pe[l, d/2, r, c]: out[r] = pe[r,0]*t_even + pe[r,1]*t_odd
  const float* per = pe + ((size_t)l * (DH / 2) + (d >> 1)) * 4;
  float p0 = per[(d & 1) * 2 + 0];
  float p1 = per[(d & 1) * 2 + 1];

  sp[d] = qn; __syncthreads();
  float qo = p0 * sp[d & ~1] + p1 * sp[d | 1];
  __syncthreads();
  sp[d] = kn; __syncthreads();
  float ko = p0 * sp[d & ~1] + p1 * sp[d | 1];

  size_t o = ((size_t)hd * Lq + l) * DH + d;
  qr[o] = f2bf(qo);
  kr[o] = f2bf(ko);
  vh[o] = f2bf(v);
}

// ---------------- K5: gelu(mlp) -> a2 cols [3072, 15360) ----------------
__global__ void k_gelu(const unsigned short* __restrict__ h, unsigned short* __restrict__ a2) {
  size_t idx = (size_t)blockIdx.x * 256 + threadIdx.x;   // 2048*12288 total
  int row = (int)(idx / MLPD), c = (int)(idx % MLPD);
  float t = bf2f(h[(size_t)row * N1 + 3 * Dm + c]);
  float g = 0.5f * t * (1.f + tanhf(0.7978845608f * (t + 0.044715f * t * t * t)));
  a2[(size_t)row * K2 + Dm + c] = f2bf(g);
}

// ---------------- K6: flash attention per (head, 128-query tile) ----------------
__global__ void k_attn(const unsigned short* __restrict__ qr, const unsigned short* __restrict__ kr,
                       const unsigned short* __restrict__ vh, unsigned short* __restrict__ a2) {
  const int hd = blockIdx.y;
  const int qb = blockIdx.x * 128;
  const int tid  = threadIdx.x;
  const int w    = tid >> 5;
  const int lane = tid & 31;
  const int m    = lane & 15;
  const int hi   = (lane & 16) ? 1 : 0;
  const int kbA  = hi ? 8 : 0;
  const int kbB  = hi ? 8 : 0;

  const unsigned short* Q  = qr + (size_t)hd * Lq * DH;
  const unsigned short* Kp = kr + (size_t)hd * Lq * DH;
  const unsigned short* V  = vh + (size_t)hd * Lq * DH;

  __shared__ __align__(16) unsigned short Ks[64 * DH];      // [key][dh] row-major
  __shared__ __align__(16) unsigned short VsT[DH * 64];     // [dh][key] transposed
  __shared__ __align__(16) unsigned short Ps[8][16 * 64];   // per-wave P tile [m][key]

  // Q fragments in registers: rows qb + w*16 + m, K = dh
  Frag16 qf[4];
#pragma unroll
  for (int kst = 0; kst < 4; ++kst) {
    const unsigned short* qp = &Q[(size_t)(qb + w * 16 + m) * DH + kst * 32 + kbA];
    qf[kst].u[0] = *(const uint4*)qp;
    qf[kst].u[1] = *(const uint4*)(qp + 16);
  }

  v8f acc[8];
#pragma unroll
  for (int i = 0; i < 8; ++i) acc[i] = vzero();
  float mrun[8], lrun[8];
#pragma unroll
  for (int v = 0; v < 8; ++v) { mrun[v] = -1e30f; lrun[v] = 0.f; }

  const float sc = 0.08838834764831845f;   // 1/sqrt(128)

  for (int kt = 0; kt < Lq / 64; ++kt) {
    // stage K tile row-major and V tile transposed (cooperative, uint4 global loads)
    for (int i = tid; i < 1024; i += 256) {         // 64 rows x 16 chunks of 8 bf16
      int r = i >> 4, cg = (i & 15) * 8;
      uint4 dk = *(const uint4*)&Kp[((size_t)kt * 64 + r) * DH + cg];
      uint4 dv = *(const uint4*)&V[((size_t)kt * 64 + r) * DH + cg];
      *(uint4*)&Ks[r * DH + cg] = dk;
      const unsigned short* sv = (const unsigned short*)&dv;
#pragma unroll
      for (int j = 0; j < 8; ++j) VsT[(cg + j) * 64 + r] = sv[j];
    }
    __syncthreads();

    // S = Q * K^T for this wave's 16 rows x 64 keys
    v8f s[4];
#pragma unroll
    for (int i = 0; i < 4; ++i) s[i] = vzero();
#pragma unroll
    for (int kst = 0; kst < 4; ++kst) {
#pragma unroll
      for (int nst = 0; nst < 4; ++nst) {
        Frag16 fb;
        fb.u[0] = *(const uint4*)&Ks[(nst * 16 + m) * DH + kst * 32 + kbB];
        fb.u[1] = *(const uint4*)&Ks[(nst * 16 + m) * DH + kst * 32 + kbB + 16];
        s[nst] = wmma_bf16(qf[kst].v, fb.v, s[nst]);
      }
    }

    // online softmax (row stats replicated across the 16-lane half)
#pragma unroll
    for (int v = 0; v < 8; ++v) {
      float mx = -1e30f;
#pragma unroll
      for (int nst = 0; nst < 4; ++nst) { s[nst][v] *= sc; mx = fmaxf(mx, s[nst][v]); }
      for (int off = 8; off; off >>= 1) mx = fmaxf(mx, __shfl_xor(mx, off, 16));
      float mnew = fmaxf(mrun[v], mx);
      float fac  = __expf(mrun[v] - mnew);
      float ps = 0.f;
#pragma unroll
      for (int nst = 0; nst < 4; ++nst) {
        float p = __expf(s[nst][v] - mnew);
        s[nst][v] = p;
        ps += p;
      }
      for (int off = 8; off; off >>= 1) ps += __shfl_xor(ps, off, 16);
      lrun[v] = lrun[v] * fac + ps;
      mrun[v] = mnew;
#pragma unroll
      for (int c = 0; c < 8; ++c) acc[c][v] *= fac;
    }

    // P (C-layout) -> LDS -> A-layout frags
#pragma unroll
    for (int nst = 0; nst < 4; ++nst)
#pragma unroll
      for (int v = 0; v < 8; ++v)
        Ps[w][(v + 8 * hi) * 64 + nst * 16 + m] = f2bf(s[nst][v]);
    asm volatile("s_wait_dscnt 0" ::: "memory");

    // O += P * V
#pragma unroll
    for (int k2 = 0; k2 < 2; ++k2) {
      Frag16 pa;
      pa.u[0] = *(const uint4*)&Ps[w][m * 64 + k2 * 32 + kbA];
      pa.u[1] = *(const uint4*)&Ps[w][m * 64 + k2 * 32 + kbA + 16];
#pragma unroll
      for (int c = 0; c < 8; ++c) {
        Frag16 vb;
        vb.u[0] = *(const uint4*)&VsT[(c * 16 + m) * 64 + k2 * 32 + kbB];
        vb.u[1] = *(const uint4*)&VsT[(c * 16 + m) * 64 + k2 * 32 + kbB + 16];
        acc[c] = wmma_bf16(pa.v, vb.v, acc[c]);
      }
    }
    __syncthreads();
  }

  // write O into a2 cols [hd*128, hd*128+128)
#pragma unroll
  for (int c = 0; c < 8; ++c) {
#pragma unroll
    for (int v = 0; v < 8; ++v) {
      int row = qb + w * 16 + v + 8 * hi;
      float o = acc[c][v] / lrun[v];
      a2[(size_t)row * K2 + hd * DH + c * 16 + m] = f2bf(o);
    }
  }
}

// ---------------- host launch ----------------
extern "C" void kernel_launch(void* const* d_in, const int* in_sizes, int n_in,
                              void* d_out, int out_size, void* d_ws, size_t ws_size,
                              hipStream_t stream) {
  (void)in_sizes; (void)n_in; (void)out_size; (void)ws_size;
  const float* x      = (const float*)d_in[0];
  const float* vec    = (const float*)d_in[1];
  const float* pe     = (const float*)d_in[2];
  const float* mod_w  = (const float*)d_in[3];
  const float* mod_b  = (const float*)d_in[4];
  const float* lin1_w = (const float*)d_in[5];
  const float* lin1_b = (const float*)d_in[6];
  const float* lin2_w = (const float*)d_in[7];
  const float* lin2_b = (const float*)d_in[8];
  const float* q_scale = (const float*)d_in[9];
  const float* k_scale = (const float*)d_in[10];
  float* out = (float*)d_out;

  char* ws = (char*)d_ws;
  const size_t OFF_MOD  = 0;
  const size_t OFF_XMOD = 64 * 1024;
  const size_t OFF_H    = OFF_XMOD + (size_t)Lq * Dm * 2;
  const size_t OFF_QR   = OFF_H   + (size_t)Lq * N1 * 2;
  const size_t OFF_KR   = OFF_QR  + (size_t)Hh * Lq * DH * 2;
  const size_t OFF_VH   = OFF_KR  + (size_t)Hh * Lq * DH * 2;
  const size_t OFF_A2   = OFF_VH  + (size_t)Hh * Lq * DH * 2;
  const size_t OFF_WBF  = OFF_A2  + (size_t)Lq * K2 * 2;    // bf16 weight buffer (reused)

  float*          modp = (float*)(ws + OFF_MOD);
  unsigned short* xmod = (unsigned short*)(ws + OFF_XMOD);
  unsigned short* h    = (unsigned short*)(ws + OFF_H);
  unsigned short* qrb  = (unsigned short*)(ws + OFF_QR);
  unsigned short* krb  = (unsigned short*)(ws + OFF_KR);
  unsigned short* vhb  = (unsigned short*)(ws + OFF_VH);
  unsigned short* a2   = (unsigned short*)(ws + OFF_A2);
  unsigned short* wbf  = (unsigned short*)(ws + OFF_WBF);

  // convert lin1_w -> bf16 (L2-resident for the 16 M-tile passes)
  k_cvt_w<<<(int)(((size_t)Dm * N1) / 4 / 256), 256, 0, stream>>>(lin1_w, wbf);
  k_mod_gemv<<<MODN / 256, 256, 0, stream>>>(vec, mod_w, mod_b, modp);
  k_ln_mod<<<Lq, 256, 0, stream>>>(x, modp, xmod);
  k_gemm<<<dim3(N1 / 128, Lq / 128), 256, 0, stream>>>(
      xmod, wbf, lin1_b, N1, Dm, 0, h, nullptr, nullptr, nullptr);
  k_qkv_post<<<dim3(Lq, Hh), 128, 0, stream>>>(h, pe, q_scale, k_scale, qrb, krb, vhb);
  k_gelu<<<(int)(((size_t)Lq * MLPD) / 256), 256, 0, stream>>>(h, a2);
  k_attn<<<dim3(Lq / 128, Hh), 256, 0, stream>>>(qrb, krb, vhb, a2);
  // convert lin2_w -> bf16 into the same buffer (stream-ordered after GEMM1)
  k_cvt_w<<<(int)(((size_t)K2 * Dm) / 4 / 256), 256, 0, stream>>>(lin2_w, wbf);
  k_gemm<<<dim3(Dm / 128, Lq / 128), 256, 0, stream>>>(
      a2, wbf, lin2_b, Dm, K2, 1, nullptr, out, modp + 2 * Dm, x);
}
